// Mamba_29557964931129
// MI455X (gfx1250) — matrix operand order repaired and checked by reading, not compiled, not run on hardware
//
#include <hip/hip_runtime.h>
#include <hip/hip_bf16.h>
#include <cstdint>
#include <cstddef>

// ---------------- model dims (fixed by the reference) ----------------
#define SEQ     2048
#define DMODEL  1024
#define DINNER  2048
#define DSTATE  16
#define DCONV   4
#define DTRANK  64
#define NLAYER  2
#define VOCABN  32000
#define XDBL_N  (DTRANK + 2 * DSTATE)   // 96

// ---------------- WMMA types ----------------
typedef __attribute__((ext_vector_type(16))) __bf16 bf16x16;
typedef __attribute__((ext_vector_type(8)))  __bf16 bf16x8;
typedef __attribute__((ext_vector_type(4)))  __bf16 bf16x4;
typedef __attribute__((ext_vector_type(8)))  float  f32x8;

union Frag16 { bf16x16 v; bf16x8 h[2]; };
union PkBf   { __bf16 b[2]; unsigned u; };

static __device__ __forceinline__ f32x8 wmma_bf16(const Frag16& a, const Frag16& b, f32x8 c)
{
    return __builtin_amdgcn_wmma_f32_16x16x32_bf16(
        false, a.v, false, b.v, (short)0, c, false, false);
}

// ---------------- GEMM: out(MxN) = A(MxK, row-major, stride lda) * W(KxN, row-major)
// epilogue: 0 = none, 1 = +bias, 2 = softplus(x + bias), 3 = +bias + residual
// Requires: M % 128 == 0, K % 64 == 0, N % 4 == 0 (N tile-padded with guards).
#define BM 128
#define BN 128
#define BK 64
#define AST 72          // ldsA row stride in bf16: 144B (16B aligned, banks staggered)
#define WST 72          // ldsW row stride in bf16 (transposed [n][k])

// one stage of global data held in registers (split-phase staging)
struct Stage {
    float4 a[8];                 // A: BMxBK, 2048 float4 / 256 thr = 8 each
    float4 w0[4], w1[4];         // W: BKxBN, k-pair loads, 4 groups each
};

static __device__ __forceinline__ void stage_global_load(
    Stage& s, const float* __restrict__ A, int lda,
    const float* __restrict__ W, int N,
    int block_m, int block_n, int k0, int tid)
{
    #pragma unroll
    for (int j = 0; j < 8; ++j) {
        int g = j * 256 + tid;
        int row = g >> 4, colq = (g & 15) << 2;
        s.a[j] = *(const float4*)&A[(size_t)(block_m + row) * lda + (k0 + colq)];
    }
    #pragma unroll
    for (int j = 0; j < 4; ++j) {
        int g  = j * 256 + tid;
        int kp = g >> 5, nq = (g & 31) << 2;   // k-pair 0..31, n quad 0..124
        int n0 = block_n + nq;
        if (n0 < N) {
            s.w0[j] = *(const float4*)&W[(size_t)(k0 + 2 * kp)     * N + n0];
            s.w1[j] = *(const float4*)&W[(size_t)(k0 + 2 * kp + 1) * N + n0];
        } else {
            s.w0[j] = float4{0.f, 0.f, 0.f, 0.f};
            s.w1[j] = float4{0.f, 0.f, 0.f, 0.f};
        }
    }
}

static __device__ __forceinline__ void stage_lds_store(
    const Stage& s, __bf16* ldsA, __bf16* ldsW, int tid)
{
    #pragma unroll
    for (int j = 0; j < 8; ++j) {
        int g = j * 256 + tid;
        int row = g >> 4, colq = (g & 15) << 2;
        bf16x4 o;
        o[0] = (__bf16)s.a[j].x; o[1] = (__bf16)s.a[j].y;
        o[2] = (__bf16)s.a[j].z; o[3] = (__bf16)s.a[j].w;
        *(bf16x4*)&ldsA[row * AST + colq] = o;          // ds_store_b64
    }
    #pragma unroll
    for (int j = 0; j < 4; ++j) {
        int g  = j * 256 + tid;
        int kp = g >> 5, nq = (g & 31) << 2;
        const float* w0a = (const float*)&s.w0[j];
        const float* w1a = (const float*)&s.w1[j];
        #pragma unroll
        for (int jj = 0; jj < 4; ++jj) {
            PkBf p;
            p.b[0] = (__bf16)w0a[jj];
            p.b[1] = (__bf16)w1a[jj];
            *(unsigned*)&ldsW[(nq + jj) * WST + 2 * kp] = p.u;   // ds_store_b32
        }
    }
}

__global__ __launch_bounds__(256) void gemm_wmma_bf16(
    const float* __restrict__ A, int lda,
    const float* __restrict__ W,
    const float* __restrict__ bias,
    const float* __restrict__ residual,
    float* __restrict__ out,
    int M, int N, int K, int epilogue)
{
    __shared__ __align__(16) __bf16 ldsA[2][BM * AST];   // [m][k], double buffered
    __shared__ __align__(16) __bf16 ldsW[2][BN * WST];   // [n][k] (transposed)

    const int tid  = threadIdx.x;
    const int wave = tid >> 5;
    const int lane = tid & 31;
    const int block_m = blockIdx.y * BM;
    const int block_n = blockIdx.x * BN;
    // 8 waves: 4 x 2 over (M,N); each wave owns a 32x64 output (2x4 WMMA tiles)
    const int wm = wave >> 1;            // 0..3
    const int wn = wave & 1;             // 0..1

    // lane -> fragment geometry (CDNA5 ISA, 16-bit layouts, wave32)
    const int r16    = lane & 15;
    const int hihalf = lane >> 4;        // 0 or 1
    const int a_kb   = hihalf * 8;       // A frag: K base 0/8, second half +16
    const int b_kb   = hihalf * 8;       // B frag: K base 0/8, second half +16

    f32x8 acc[2][4] = {};

    Stage st;
    stage_global_load(st, A, lda, W, N, block_m, block_n, 0, tid);
    stage_lds_store(st, ldsA[0], ldsW[0], tid);
    __syncthreads();

    int buf = 0;
    for (int k0 = 0; k0 < K; k0 += BK) {
        const bool has_next = (k0 + BK) < K;
        if (has_next)      // global loads for next stage fly under this stage's WMMAs
            stage_global_load(st, A, lda, W, N, block_m, block_n, k0 + BK, tid);
        if (k0 + 2 * BK < K)   // pull stage-after-next into GL2 (global_prefetch_b8)
            __builtin_prefetch(&W[(size_t)(k0 + 2 * BK + (tid >> 3)) * N
                                  + block_n + (tid & 7) * 16], 0, 1);

        const __bf16* lA = ldsA[buf];
        const __bf16* lW = ldsW[buf];
        #pragma unroll
        for (int kk = 0; kk < BK; kk += 32) {
            Frag16 a[2], b[4];
            #pragma unroll
            for (int ti = 0; ti < 2; ++ti) {
                const __bf16* ap = &lA[(wm * 32 + ti * 16 + r16) * AST + kk];
                a[ti].h[0] = *(const bf16x8*)(ap + a_kb);
                a[ti].h[1] = *(const bf16x8*)(ap + a_kb + 16);
            }
            #pragma unroll
            for (int tj = 0; tj < 4; ++tj) {
                const __bf16* bp = &lW[(wn * 64 + tj * 16 + r16) * WST + kk];
                b[tj].h[0] = *(const bf16x8*)(bp + b_kb);
                b[tj].h[1] = *(const bf16x8*)(bp + b_kb + 16);
            }
            #pragma unroll
            for (int ti = 0; ti < 2; ++ti)
                #pragma unroll
                for (int tj = 0; tj < 4; ++tj)
                    acc[ti][tj] = wmma_bf16(a[ti], b[tj], acc[ti][tj]);
        }

        if (has_next) {
            stage_lds_store(st, ldsA[buf ^ 1], ldsW[buf ^ 1], tid);
            buf ^= 1;
        }
        __syncthreads();
    }

    // ---- epilogue + store: elem i -> row = base + (lane>>4)*8 + i, col = lane&15
    const int row_base = block_m + wm * 32 + hihalf * 8;
    const int col_base = block_n + wn * 64 + r16;
    #pragma unroll
    for (int tj = 0; tj < 4; ++tj) {
        int c = col_base + tj * 16;
        if (c >= N) continue;
        float bv = (epilogue >= 1) ? bias[c] : 0.0f;
        #pragma unroll
        for (int ti = 0; ti < 2; ++ti) {
            int r0 = row_base + ti * 16;
            #pragma unroll
            for (int i = 0; i < 8; ++i) {
                int r = r0 + i;
                float v = acc[ti][tj][i] + bv;
                if (epilogue == 2) {                   // softplus
                    v = (v > 20.0f) ? v : __logf(1.0f + __expf(v));
                } else if (epilogue == 3) {            // residual add
                    v += residual[(size_t)r * N + c];
                }
                out[(size_t)r * N + c] = v;
            }
        }
    }
}

// ---------------- embedding gather ----------------
__global__ __launch_bounds__(256) void embed_kernel(
    const int* __restrict__ ids, const float* __restrict__ emb,
    float* __restrict__ h)
{
    int l = blockIdx.x;
    int tok = ids[l];
    for (int j = threadIdx.x; j < DMODEL; j += 256)
        h[(size_t)l * DMODEL + j] = emb[(size_t)tok * DMODEL + j];
}

// ---------------- rmsnorm over rows of length d ----------------
__global__ __launch_bounds__(256) void rmsnorm_kernel(
    const float* __restrict__ x, const float* __restrict__ scale,
    float* __restrict__ out, int d)
{
    __shared__ float red[256];
    int row = blockIdx.x;
    const float* xr = x + (size_t)row * d;
    float ss = 0.0f;
    for (int j = threadIdx.x; j < d; j += 256) { float v = xr[j]; ss += v * v; }
    red[threadIdx.x] = ss;
    __syncthreads();
    for (int off = 128; off > 0; off >>= 1) {
        if (threadIdx.x < off) red[threadIdx.x] += red[threadIdx.x + off];
        __syncthreads();
    }
    float inv = rsqrtf(red[0] / (float)d + 1e-6f);
    for (int j = threadIdx.x; j < d; j += 256)
        out[(size_t)row * d + j] = xr[j] * inv * scale[j];
}

// ---------------- repo-quirk depthwise conv over d_inner axis + SiLU ----------
// xc[l,d] = silu( sum_{k=0..3} x[l, d+k-1] * w[k, l] + b[l] ),  x = xr[:, :DINNER]
__global__ __launch_bounds__(256) void conv_silu_kernel(
    const float* __restrict__ xr, const float* __restrict__ w,
    const float* __restrict__ b, float* __restrict__ xc)
{
    int idx = blockIdx.x * 256 + threadIdx.x;     // l*DINNER + d
    int l = idx >> 11;                            // /DINNER
    int d = idx & (DINNER - 1);
    float acc = b[l];
    #pragma unroll
    for (int k = 0; k < DCONV; ++k) {
        int dd = d + k - 1;                       // SAME pad: low=1, high=2
        float xv = (dd >= 0 && dd < DINNER) ? xr[(size_t)l * (2 * DINNER) + dd] : 0.0f;
        acc += xv * w[k * DINNER + l];
    }
    xc[idx] = acc / (1.0f + __expf(-acc));        // silu
}

// ---------------- selective scan: 1 lane = 1 channel, recurrence over l ------
__global__ __launch_bounds__(256) void scan_kernel(
    const float* __restrict__ u, const float* __restrict__ delta,
    const float* __restrict__ xdbl, const float* __restrict__ A_log,
    const float* __restrict__ Dp, float* __restrict__ y)
{
    int d = blockIdx.x * 256 + threadIdx.x;       // 0..DINNER-1
    float negA[DSTATE], s[DSTATE];
    #pragma unroll
    for (int n = 0; n < DSTATE; ++n) {
        negA[n] = -__expf(A_log[(size_t)d * DSTATE + n]);   // A = -exp(A_log)
        s[n] = 0.0f;
    }
    float Dd = Dp[d];
    for (int l = 0; l < SEQ; ++l) {
        float dl = delta[(size_t)l * DINNER + d];
        float ul = u[(size_t)l * DINNER + d];
        const float* bc = xdbl + (size_t)l * XDBL_N;        // wave-uniform
        float acc = 0.0f;
        #pragma unroll
        for (int n = 0; n < DSTATE; ++n) {
            float dA = __expf(dl * negA[n]);
            s[n] = dA * s[n] + (dl * bc[DTRANK + n]) * ul;
            acc += s[n] * bc[DTRANK + DSTATE + n];
        }
        y[(size_t)l * DINNER + d] = acc + ul * Dd;
    }
}

// ---------------- gating: y *= silu(res) ----------------
__global__ __launch_bounds__(256) void gate_kernel(
    float* __restrict__ y, const float* __restrict__ xr)
{
    int idx = blockIdx.x * 256 + threadIdx.x;     // l*DINNER + d
    int l = idx >> 11;
    int d = idx & (DINNER - 1);
    float r = xr[(size_t)l * (2 * DINNER) + DINNER + d];
    y[idx] *= r / (1.0f + __expf(-r));
}

// ---------------- host-side launch ----------------
static inline void launch_gemm(const float* A, int lda, const float* W,
                               const float* bias, const float* residual,
                               float* out, int M, int N, int K, int epi,
                               hipStream_t stream)
{
    dim3 grid((N + BN - 1) / BN, M / BM);
    gemm_wmma_bf16<<<grid, 256, 0, stream>>>(A, lda, W, bias, residual, out,
                                             M, N, K, epi);
}

extern "C" void kernel_launch(void* const* d_in, const int* in_sizes, int n_in,
                              void* d_out, int out_size, void* d_ws, size_t ws_size,
                              hipStream_t stream)
{
    const int*   ids    = (const int*)  d_in[0];
    const float* emb    = (const float*)d_in[1];
    const float* nscale = (const float*)d_in[2];
    const float* in_w   = (const float*)d_in[3];
    const float* in_b   = (const float*)d_in[4];
    const float* conv_w = (const float*)d_in[5];
    const float* conv_b = (const float*)d_in[6];
    const float* xp_w   = (const float*)d_in[7];
    const float* dt_w   = (const float*)d_in[8];
    const float* dt_b   = (const float*)d_in[9];
    const float* out_w  = (const float*)d_in[10];
    const float* out_b  = (const float*)d_in[11];
    const float* A_log  = (const float*)d_in[12];
    const float* Dp     = (const float*)d_in[13];
    const float* normf  = (const float*)d_in[14];
    const float* lm_w   = (const float*)d_in[15];
    float* out = (float*)d_out;

    // workspace carve-out (floats)
    float* h     = (float*)d_ws;                          // SEQ*DMODEL
    float* u     = h     + (size_t)SEQ * DMODEL;          // SEQ*DMODEL
    float* xr    = u     + (size_t)SEQ * DMODEL;          // SEQ*2*DINNER
    float* xc    = xr    + (size_t)SEQ * 2 * DINNER;      // SEQ*DINNER
    float* xdbl  = xc    + (size_t)SEQ * DINNER;          // SEQ*XDBL_N
    float* delta = xdbl  + (size_t)SEQ * XDBL_N;          // SEQ*DINNER
    float* y     = delta + (size_t)SEQ * DINNER;          // SEQ*DINNER

    embed_kernel<<<SEQ, 256, 0, stream>>>(ids, emb, h);

    for (int L = 0; L < NLAYER; ++L) {
        // u = rmsnorm(h)
        rmsnorm_kernel<<<SEQ, 256, 0, stream>>>(h, nscale + (size_t)L * DMODEL, u, DMODEL);
        // xr = u @ in_w + in_b
        launch_gemm(u, DMODEL, in_w + (size_t)L * DMODEL * 2 * DINNER,
                    in_b + (size_t)L * 2 * DINNER, nullptr, xr,
                    SEQ, 2 * DINNER, DMODEL, /*epi=*/1, stream);
        // xc = silu(depthwise-conv-over-d(xr[:, :DINNER]))
        conv_silu_kernel<<<(SEQ * DINNER) / 256, 256, 0, stream>>>(
            xr, conv_w + (size_t)L * DCONV * DINNER, conv_b + (size_t)L * DINNER, xc);
        // xdbl = u @ xp_w   (N=96: tile guards handle the partial block)
        launch_gemm(u, DMODEL, xp_w + (size_t)L * DMODEL * XDBL_N,
                    nullptr, nullptr, xdbl, SEQ, XDBL_N, DMODEL, /*epi=*/0, stream);
        // delta = softplus(xdbl[:, :DTRANK] @ dt_w + dt_b)
        launch_gemm(xdbl, XDBL_N, dt_w + (size_t)L * DTRANK * DINNER,
                    dt_b + (size_t)L * DINNER, nullptr, delta,
                    SEQ, DINNER, DTRANK, /*epi=*/2, stream);
        // y = selective_scan(xc, delta, A, B, C, D)
        scan_kernel<<<DINNER / 256, 256, 0, stream>>>(
            xc, delta, xdbl, A_log + (size_t)L * DINNER * DSTATE,
            Dp + (size_t)L * DINNER, y);
        // y *= silu(res)
        gate_kernel<<<(SEQ * DINNER) / 256, 256, 0, stream>>>(y, xr);
        // h = y @ out_w + out_b + h
        launch_gemm(y, DINNER, out_w + (size_t)L * DINNER * DMODEL,
                    out_b + (size_t)L * DMODEL, h, h,
                    SEQ, DMODEL, DINNER, /*epi=*/3, stream);
    }

    // final norm + lm_head
    rmsnorm_kernel<<<SEQ, 256, 0, stream>>>(h, normf, u, DMODEL);
    launch_gemm(u, DMODEL, lm_w, nullptr, nullptr, out,
                SEQ, VOCABN, DMODEL, /*epi=*/0, stream);
}
